// MultiHeadCrossAttention_15779709845615
// MI455X (gfx1250) — compile-verified, matrix operand order checked by reading, not run on hardware
//
#include <hip/hip_runtime.h>

typedef __attribute__((ext_vector_type(16))) __bf16 v16bf;
typedef __attribute__((ext_vector_type(8)))  float  v8f;

union FragU { v16bf v; uint4 q[2]; };

// Load one 16-bit WMMA A/B fragment for this lane.
// p must already point at: base + row*ld + k0 + (lane>>4)*8   (elements)
// Per-lane data = elements [0..7] and [16..23] relative to p (ISA 16-bit A layout).
__device__ __forceinline__ v16bf load_frag(const unsigned short* p) {
    FragU f;
    f.q[0] = *reinterpret_cast<const uint4*>(p);
    f.q[1] = *reinterpret_cast<const uint4*>(p + 16);
    return f.v;
}

__device__ __forceinline__ unsigned short f32_to_bf16(float f) {
    unsigned int u = __float_as_uint(f);
    unsigned int r = u + 0x7FFFu + ((u >> 16) & 1u);   // round-to-nearest-even
    return (unsigned short)(r >> 16);
}

// Async copy 16 bytes global -> LDS for this lane (ASYNCcnt-tracked).
__device__ __forceinline__ void async_g2l_b128(const unsigned short* gptr, void* lptr) {
    unsigned int       l = (unsigned int)(uintptr_t)lptr;        // LDS byte offset
    unsigned long long g = (unsigned long long)(uintptr_t)gptr;  // 64-bit global addr
    asm volatile("global_load_async_to_lds_b128 %0, %1, off" :: "v"(l), "v"(g) : "memory");
}
__device__ __forceinline__ void wait_async0() {
    asm volatile("s_wait_asynccnt 0" ::: "memory");
}

#define WMMA_BF16(a, b, c) \
    __builtin_amdgcn_wmma_f32_16x16x32_bf16(false, (a), false, (b), (short)0, (c), false, false)

// ---------------------------------------------------------------- pre-passes
__global__ void cvt_f32_bf16(const float* __restrict__ in,
                             unsigned short* __restrict__ out, int n) {
    int i = blockIdx.x * 256 + threadIdx.x;
    if (i < n) out[i] = f32_to_bf16(in[i]);
}

// W[K,N] fp32 -> Wt[N,K] bf16
__global__ void transpose_f32_bf16(const float* __restrict__ W,
                                   unsigned short* __restrict__ Wt, int K, int N) {
    int i = blockIdx.x * 256 + threadIdx.x;
    if (i < K * N) {
        int k = i / N, n = i - k * N;
        Wt[(size_t)n * K + k] = f32_to_bf16(W[i]);
    }
}

// ------------------------------------------------------------------- GEMM
// C[M,N] fp32 = A[M,K]bf16 @ Bt[N,K]bf16^T + bias[N]
// block = 128 (4 waves); grid = (M/16, N/256); wave computes 16x64.
// A tile (16x32, shared by the 4 waves) is double-buffered in LDS via
// global_load_async_to_lds_b128 (waves 0-1 issue; prefetch chunk c+1 during chunk c).
__global__ void __launch_bounds__(128) gemm_bf16_bias(
    const unsigned short* __restrict__ A, const unsigned short* __restrict__ Bt,
    const float* __restrict__ bias, float* __restrict__ C,
    int M, int N, int K) {
    __shared__ __align__(16) unsigned short aT[2][16][40];   // 32 used + pad

    const int tid  = threadIdx.x;
    const int lane = tid & 31;
    const int wv   = tid >> 5;
    const int hl   = lane >> 4;          // 0/1 half selector
    const int l16  = lane & 15;
    const int m0 = blockIdx.x * 16;
    const int n0 = blockIdx.y * 256 + wv * 64;

    v8f acc0 = {}, acc1 = {}, acc2 = {}, acc3 = {};
    const unsigned short* brow = Bt + (size_t)(n0 + l16) * K + hl * 8;
    const size_t tstep = (size_t)16 * K;

    const int  arow_i = tid >> 2;            // 0..15 (loader rows, tid<64)
    const int  aseg   = (tid & 3) * 8;       // k element offset within chunk
    const unsigned short* aSrc = A + (size_t)(m0 + arow_i) * K + aseg;
    const bool loader = (wv < 2);            // wave-uniform

    // prologue: stage chunk 0
    if (loader) async_g2l_b128(aSrc, &aT[0][arow_i][aseg]);
    wait_async0();
    __syncthreads();

    const int NK = K >> 5;
    for (int kc = 0; kc < NK; ++kc) {
        const int buf = kc & 1;
        if (kc + 1 < NK && loader)                     // prefetch next chunk
            async_g2l_b128(aSrc + ((kc + 1) << 5), &aT[buf ^ 1][arow_i][aseg]);

        const int k0 = kc << 5;
        v16bf a  = load_frag(&aT[buf][l16][hl * 8]);   // ds_load_b128 x2
        v16bf b0 = load_frag(brow + k0);
        v16bf b1 = load_frag(brow + tstep     + k0);
        v16bf b2 = load_frag(brow + 2 * tstep + k0);
        v16bf b3 = load_frag(brow + 3 * tstep + k0);
        acc0 = WMMA_BF16(a, b0, acc0);
        acc1 = WMMA_BF16(a, b1, acc1);
        acc2 = WMMA_BF16(a, b2, acc2);
        acc3 = WMMA_BF16(a, b3, acc3);

        wait_async0();                                  // prefetch landed
        __syncthreads();
    }

    const int rb = m0 + hl * 8;
    const float bi0 = bias[n0 + l16];
    const float bi1 = bias[n0 + 16 + l16];
    const float bi2 = bias[n0 + 32 + l16];
    const float bi3 = bias[n0 + 48 + l16];
#pragma unroll
    for (int r = 0; r < 8; ++r) {
        float* crow = C + (size_t)(rb + r) * N;
        crow[n0 + l16]      = acc0[r] + bi0;
        crow[n0 + 16 + l16] = acc1[r] + bi1;
        crow[n0 + 32 + l16] = acc2[r] + bi2;
        crow[n0 + 48 + l16] = acc3[r] + bi3;
    }
}

// --------------------------------------------------------------- LayerNorm
// one block of 256 per row of Y[4096,1024]; outputs bf16.
// vmode==0: out[row,col] row-major.  vmode==1 (V): out[((b*16+h)*64+d), s] (transposed per head).
__global__ void __launch_bounds__(256) layernorm_bf16(
    const float* __restrict__ Y, const float* __restrict__ g,
    const float* __restrict__ be, unsigned short* __restrict__ out, int vmode) {
    __shared__ float red[256];
    __shared__ float stats[2];
    const int row = blockIdx.x, tid = threadIdx.x;

    float v[4]; float s = 0.f;
#pragma unroll
    for (int i = 0; i < 4; ++i) { v[i] = Y[(size_t)row * 1024 + tid + i * 256]; s += v[i]; }
    red[tid] = s; __syncthreads();
    for (int o = 128; o > 0; o >>= 1) { if (tid < o) red[tid] += red[tid + o]; __syncthreads(); }
    if (tid == 0) stats[0] = red[0] * (1.f / 1024.f);
    __syncthreads();
    const float mean = stats[0];
    float s2 = 0.f;
#pragma unroll
    for (int i = 0; i < 4; ++i) { float d = v[i] - mean; s2 += d * d; }
    __syncthreads();
    red[tid] = s2; __syncthreads();
    for (int o = 128; o > 0; o >>= 1) { if (tid < o) red[tid] += red[tid + o]; __syncthreads(); }
    if (tid == 0) stats[1] = rsqrtf(red[0] * (1.f / 1024.f) + 1e-5f);
    __syncthreads();
    const float rstd = stats[1];
#pragma unroll
    for (int i = 0; i < 4; ++i) {
        int col = tid + i * 256;
        unsigned short hb = f32_to_bf16((v[i] - mean) * rstd * g[col] + be[col]);
        if (vmode == 0) {
            out[(size_t)row * 1024 + col] = hb;
        } else {
            int b = row >> 11, sI = row & 2047;
            int h = col >> 6,  d  = col & 63;
            out[((size_t)((b * 16 + h) * 64 + d)) * 2048 + sI] = hb;
        }
    }
}

// --------------------------------------------------------------- Attention
// grid = (S1/16=128, H=16, B=2), block = 256 (8 waves).
// qb,kb: [B*S,1024] bf16 row-major; vt: [B,H,64,S2] bf16; ao: [B*S,1024] bf16.
#define CHUNK 512
#define LDSK  528   // padded row stride (multiple of 8 elems = 16B aligned)

__global__ void __launch_bounds__(256) attention_kernel(
    const unsigned short* __restrict__ qb, const unsigned short* __restrict__ kb,
    const unsigned short* __restrict__ vt, unsigned short* __restrict__ ao) {
    __shared__ float sc[16][CHUNK];                        // raw scores (chunk)
    __shared__ __align__(16) unsigned short pb[16][LDSK];  // bf16 probs (chunk)
    __shared__ __align__(16) unsigned short qt[16][72];    // staged Q tile (16x64 + pad)
    __shared__ float mrow[16], lrow[16], alpha[16];
    __shared__ float pred[16][16];

    const int tid  = threadIdx.x;
    const int lane = tid & 31, wv = tid >> 5;
    const int hl = lane >> 4, l16 = lane & 15;
    const int b = blockIdx.z, h = blockIdx.y, it = blockIdx.x;

    if (tid < 16) { mrow[tid] = -1e30f; lrow[tid] = 0.f; }

    // stage Q tile once via async DMA (waves 0-3, wave-uniform)
    if (wv < 4) {
        const int qr = tid >> 3;          // 0..15
        const int qs = (tid & 7) * 8;     // 0..56
        async_g2l_b128(qb + (size_t)(b * 2048 + it * 16 + qr) * 1024 + h * 64 + qs,
                       &qt[qr][qs]);
    }
    wait_async0();

    v8f o0 = {};  // O accumulator (waves 0..3 own d-tiles 0..3)

    const unsigned short* kbase0 = kb + (size_t)(b * 2048) * 1024 + h * 64 + hl * 8;
    const unsigned short* vbase  = vt + (size_t)((b * 16 + h) * 64 + wv * 16 + l16) * 2048 + hl * 8;
    __syncthreads();

    for (int c = 0; c < 4; ++c) {
        const int j0 = c * CHUNK;
        // ---- phase A: scores chunk -> LDS (scaled by 1/sqrt(64))
        for (int jj = wv; jj < 32; jj += 8) {
            v8f s = {};
            const unsigned short* kb2 = kbase0 + (size_t)(j0 + jj * 16 + l16) * 1024;
            v16bf a0 = load_frag(&qt[l16][hl * 8]);
            v16bf b0 = load_frag(kb2);
            s = WMMA_BF16(a0, b0, s);
            v16bf a1 = load_frag(&qt[l16][32 + hl * 8]);
            v16bf b1 = load_frag(kb2 + 32);
            s = WMMA_BF16(a1, b1, s);
#pragma unroll
            for (int r = 0; r < 8; ++r)
                sc[r + hl * 8][jj * 16 + l16] = s[r] * 0.125f;
        }
        __syncthreads();

        // ---- phase B: online softmax update (16 rows x 16 segments of 32)
        const int row = tid >> 4, seg = tid & 15;
        float mx = -1e30f;
        for (int t = 0; t < 32; ++t) mx = fmaxf(mx, sc[row][seg * 32 + t]);
        pred[row][seg] = mx;
        __syncthreads();
        if (seg == 0) {
            float m = mrow[row];
            for (int t = 0; t < 16; ++t) m = fmaxf(m, pred[row][t]);
            alpha[row] = __expf(mrow[row] - m);
            mrow[row]  = m;
        }
        __syncthreads();
        const float mr = mrow[row];
        float ssum = 0.f;
        for (int t = 0; t < 32; ++t) {
            float e = __expf(sc[row][seg * 32 + t] - mr);
            pb[row][seg * 32 + t] = f32_to_bf16(e);
            ssum += e;
        }
        pred[row][seg] = ssum;
        __syncthreads();
        if (seg == 0) {
            float s2 = 0.f;
            for (int t = 0; t < 16; ++t) s2 += pred[row][t];
            lrow[row] = lrow[row] * alpha[row] + s2;
        }
        __syncthreads();

        // ---- phase C: O = O*alpha + P @ V_chunk   (waves 0..3, uniform branch)
        if (wv < 4) {
#pragma unroll
            for (int r = 0; r < 8; ++r) o0[r] *= alpha[r + hl * 8];
            for (int ks = 0; ks < CHUNK; ks += 32) {
                v16bf a  = load_frag(&pb[l16][ks + hl * 8]);   // probs from LDS
                v16bf bb = load_frag(vbase + j0 + ks);         // V^T from global
                o0 = WMMA_BF16(a, bb, o0);
            }
        }
        __syncthreads();
    }

    if (wv < 4) {
#pragma unroll
        for (int r = 0; r < 8; ++r) {
            const int rowi = r + hl * 8;
            float o = o0[r] / lrow[rowi];
            ao[(size_t)(b * 2048 + it * 16 + rowi) * 1024 + h * 64 + wv * 16 + l16] =
                f32_to_bf16(o);
        }
    }
}

// ------------------------------------------------------------------ launch
extern "C" void kernel_launch(void* const* d_in, const int* in_sizes, int n_in,
                              void* d_out, int out_size, void* d_ws, size_t ws_size,
                              hipStream_t stream) {
    (void)in_sizes; (void)n_in; (void)out_size; (void)ws_size;

    const float* x1   = (const float*)d_in[0];
    const float* x2   = (const float*)d_in[1];
    const float* Wq   = (const float*)d_in[2];
    const float* bq   = (const float*)d_in[3];
    const float* Wk   = (const float*)d_in[4];
    const float* bk   = (const float*)d_in[5];
    const float* Wv   = (const float*)d_in[6];
    const float* bv   = (const float*)d_in[7];
    const float* gq   = (const float*)d_in[8];
    const float* betq = (const float*)d_in[9];
    const float* gk   = (const float*)d_in[10];
    const float* betk = (const float*)d_in[11];
    const float* gv   = (const float*)d_in[12];
    const float* betv = (const float*)d_in[13];
    const float* Wo   = (const float*)d_in[14];
    const float* bo   = (const float*)d_in[15];
    float* out = (float*)d_out;

    const int M = 4096;              // B*S = 2*2048
    const int D = 1024;
    const size_t actB  = (size_t)M * D * 2;   // bf16 activation buffer
    const size_t wB    = (size_t)D * D * 2;   // bf16 weight buffer
    const size_t yB    = (size_t)M * D * 4;   // fp32 scratch

    char* w = (char*)d_ws;
    size_t off = 0;
    auto alloc = [&](size_t bytes) { void* p = w + off; off += (bytes + 255) & ~(size_t)255; return p; };

    unsigned short* x1b = (unsigned short*)alloc(actB);
    unsigned short* x2b = (unsigned short*)alloc(actB);
    unsigned short* WqT = (unsigned short*)alloc(wB);
    unsigned short* WkT = (unsigned short*)alloc(wB);
    unsigned short* WvT = (unsigned short*)alloc(wB);
    unsigned short* WoT = (unsigned short*)alloc(wB);
    float*          Y   = (float*)alloc(yB);
    unsigned short* qbf = (unsigned short*)alloc(actB);
    unsigned short* kbf = (unsigned short*)alloc(actB);
    unsigned short* vtb = (unsigned short*)alloc(actB);
    unsigned short* aob = (unsigned short*)alloc(actB);

    const int nAct = M * D;
    cvt_f32_bf16<<<nAct / 256, 256, 0, stream>>>(x1, x1b, nAct);
    cvt_f32_bf16<<<nAct / 256, 256, 0, stream>>>(x2, x2b, nAct);
    const int nW = D * D;
    transpose_f32_bf16<<<nW / 256, 256, 0, stream>>>(Wq, WqT, D, D);
    transpose_f32_bf16<<<nW / 256, 256, 0, stream>>>(Wk, WkT, D, D);
    transpose_f32_bf16<<<nW / 256, 256, 0, stream>>>(Wv, WvT, D, D);
    transpose_f32_bf16<<<nW / 256, 256, 0, stream>>>(Wo, WoT, D, D);

    dim3 gg(M / 16, D / 256);
    gemm_bf16_bias<<<gg, 128, 0, stream>>>(x1b, WqT, bq, Y, M, D, D);
    layernorm_bf16<<<M, 256, 0, stream>>>(Y, gq, betq, qbf, 0);
    gemm_bf16_bias<<<gg, 128, 0, stream>>>(x2b, WkT, bk, Y, M, D, D);
    layernorm_bf16<<<M, 256, 0, stream>>>(Y, gk, betk, kbf, 0);
    gemm_bf16_bias<<<gg, 128, 0, stream>>>(x2b, WvT, bv, Y, M, D, D);
    layernorm_bf16<<<M, 256, 0, stream>>>(Y, gv, betv, vtb, 1);

    attention_kernel<<<dim3(128, 16, 2), 256, 0, stream>>>(qbf, kbf, vtb, aob);

    gemm_bf16_bias<<<gg, 128, 0, stream>>>(aob, WoT, bo, out, M, D, D);
}
